// Head_44933947851067
// MI455X (gfx1250) — matrix operand order repaired and checked by reading, not compiled, not run on hardware
//
#include <hip/hip_runtime.h>
#include <hip/hip_bf16.h>

// ---- types for WMMA fragments ------------------------------------------------
typedef __attribute__((ext_vector_type(16))) __bf16          v16bf;
typedef __attribute__((ext_vector_type(8)))  float           v8f;
typedef __attribute__((ext_vector_type(4)))  float           f4;
typedef __attribute__((ext_vector_type(16))) unsigned short  us16;
typedef __attribute__((ext_vector_type(8)))  unsigned short  us8;

#define S_LEN  4096
#define BATCH  4
#define DMODEL 1024
#define HDIM   64
#define ROWS   (BATCH * S_LEN)   // 16384

__device__ __forceinline__ unsigned short bf16bits(float f) {
  __bf16 h = (__bf16)f;                       // v_cvt_pk_bf16_f32 path
  return __builtin_bit_cast(unsigned short, h);
}

// B fragment: 16 contiguous K values per lane (32B -> 2x b128)
__device__ __forceinline__ v16bf ld_bfrag(const unsigned short* p) {
  us16 u = *reinterpret_cast<const us16*>(p);
  return __builtin_bit_cast(v16bf, u);
}

// A fragment from two contiguous 8-element chunks
__device__ __forceinline__ v16bf ld_afrag(const unsigned short* p0,
                                          const unsigned short* p1) {
  us8 a = *reinterpret_cast<const us8*>(p0);
  us8 b = *reinterpret_cast<const us8*>(p1);
  us16 u;
#pragma unroll
  for (int i = 0; i < 8; ++i) { u[i] = a[i]; u[8 + i] = b[i]; }
  return __builtin_bit_cast(v16bf, u);
}

__device__ __forceinline__ v8f wmma_bf16(v16bf a, v16bf b, v8f c) {
  return __builtin_amdgcn_wmma_f32_16x16x32_bf16(false, a, false, b,
                                                 (short)0, c, false, false);
}

// low 32 bits of a generic shared pointer == LDS byte offset
__device__ __forceinline__ unsigned ldsoff(const void* p) {
  return (unsigned)(unsigned long long)p;
}

// CDNA5 async global->LDS copy (16 bytes per lane), tracked by ASYNCcnt
__device__ __forceinline__ void async_copy16(unsigned lds, const void* g) {
  asm volatile("global_load_async_to_lds_b128 %0, %1, off"
               :: "v"(lds), "v"(g) : "memory");
}
__device__ __forceinline__ void wait_async0() {
  asm volatile("s_wait_asynccnt 0x0" ::: "memory");
}

// ============================================================================
// Kernel 1: transpose+convert weights -> bf16 Wt[3*HDIM rows][DMODEL]
// ============================================================================
__global__ void wt_kernel(const float* __restrict__ Wq,
                          const float* __restrict__ Wk,
                          const float* __restrict__ Wv,
                          unsigned short* __restrict__ Wt) {
  int i = blockIdx.x * 256 + threadIdx.x;       // 3*64*1024 = 196608 total
  if (i >= 3 * HDIM * DMODEL) return;
  int w   = i / (HDIM * DMODEL);
  int rem = i % (HDIM * DMODEL);
  int h   = rem / DMODEL;
  int kk  = rem % DMODEL;
  const float* W = (w == 0) ? Wq : ((w == 1) ? Wk : Wv);
  Wt[i] = bf16bits(W[kk * HDIM + h]);
}

// ============================================================================
// Kernel 2: fused QKV projection. 8 waves/block, 128 rows/block.
// Weight K-slices async-staged into double-buffered LDS; A (x) double-buffered
// in registers. q pre-scaled by 1/sqrt(64); v stored transposed vT[B][H][S].
// ============================================================================
__global__ __launch_bounds__(256) void qkv_kernel(
    const float* __restrict__ x, const unsigned short* __restrict__ Wt,
    unsigned short* __restrict__ q, unsigned short* __restrict__ k,
    unsigned short* __restrict__ vT) {
  // K-slice of all 192 weight rows x 32 K values, double-buffered (24 KB)
  __shared__ __attribute__((aligned(16))) unsigned short lW[2][192 * 32];

  const int lane  = threadIdx.x & 31;
  const int wave  = threadIdx.x >> 5;
  const int half  = lane >> 4;          // 0: lanes 0-15, 1: lanes 16-31
  const int l16   = lane & 15;
  const int r0    = blockIdx.x * 128 + wave * 16;
  const int mrow  = r0 + l16;           // A-matrix row for this lane
  const int ksh   = half ? 16 : 0;      // A K offsets 0..15 / 16..31 (this code's contiguous k order)

  // async-stage weight slice [192 rows x 32 K] for K-chunk k0 into lW[buf]
  auto stageW = [&](int buf, int k0) {
#pragma unroll
    for (int j = 0; j < 3; ++j) {                     // 768 chunks / 256 thr
      int c   = threadIdx.x + j * 256;
      int row = c >> 2;
      int off = (c & 3) * 8;
      async_copy16(ldsoff(&lW[buf][row * 32 + off]),
                   Wt + (size_t)row * DMODEL + k0 + off);
    }
  };

  f4 araw[2][4];                                      // raw fp32 A double buffer
  auto loadA = [&](int buf, int k0) {
    const float* xp = x + (size_t)mrow * DMODEL + k0 + ksh;
    araw[buf][0] = *(const f4*)(xp);
    araw[buf][1] = *(const f4*)(xp + 4);
    araw[buf][2] = *(const f4*)(xp + 8);
    araw[buf][3] = *(const f4*)(xp + 12);
  };

  v8f acc[12];
  v8f zero = {0.f, 0.f, 0.f, 0.f, 0.f, 0.f, 0.f, 0.f};
#pragma unroll
  for (int i = 0; i < 12; ++i) acc[i] = zero;

  stageW(0, 0);
  loadA(0, 0);
  wait_async0();
  __syncthreads();

#pragma unroll 2                                      // keeps buf indices const
  for (int k0 = 0; k0 < DMODEL; k0 += 32) {
    const int buf = (k0 >> 5) & 1;
    const bool more = (k0 + 32 < DMODEL);
    if (more) { stageW(buf ^ 1, k0 + 32); loadA(buf ^ 1, k0 + 32); }

    // convert current A chunk (v_cvt_pk_bf16_f32)
    v16bf a;
#pragma unroll
    for (int j = 0; j < 4; ++j)
#pragma unroll
      for (int e = 0; e < 4; ++e) a[j * 4 + e] = (__bf16)araw[buf][j][e];

    // 12 WMMAs, B fragments from LDS (fragment i -> weight row i*16+l16)
#pragma unroll
    for (int i = 0; i < 12; ++i) {
      const unsigned short* bp = &lW[buf][(i * 16 + l16) * 32 + (half ? 16 : 0)];
      acc[i] = wmma_bf16(a, ld_bfrag(bp), acc[i]);
    }

    if (more) { wait_async0(); __syncthreads(); }
  }

  const int bb   = r0 >> 12;            // batch  (r0 / S_LEN)
  const int srow = r0 & (S_LEN - 1);    // seq row within batch
#pragma unroll
  for (int t = 0; t < 4; ++t) {
    v8f cq = acc[t], ck = acc[4 + t], cv = acc[8 + t];
#pragma unroll
    for (int r = 0; r < 8; ++r) {
      int row = r0 + r + half * 8;
      int col = t * 16 + l16;
      q[(size_t)row * HDIM + col] = bf16bits(cq[r] * 0.125f);  // fold 1/sqrt(64)
      k[(size_t)row * HDIM + col] = bf16bits(ck[r]);
      vT[((size_t)bb * HDIM + col) * S_LEN + (srow + r + half * 8)] =
          bf16bits(cv[r]);
    }
  }
}

// ============================================================================
// Kernel 3: causal flash attention. Block = 64 q-rows of one batch, 4 waves.
// Double-buffered LDS tiles staged with async global->LDS copies.
// ============================================================================
__global__ __launch_bounds__(128) void flash_kernel(
    const unsigned short* __restrict__ q, const unsigned short* __restrict__ k,
    const unsigned short* __restrict__ vT, float* __restrict__ out) {
  __shared__ __attribute__((aligned(16))) unsigned short lK[2][32 * HDIM]; // [key][h]
  __shared__ __attribute__((aligned(16))) unsigned short lV[2][HDIM * 32]; // [h][key]
  __shared__ __attribute__((aligned(16))) unsigned short lP[4][16 * 32];   // P staging

  const int lane  = threadIdx.x & 31;
  const int wave  = threadIdx.x >> 5;
  const int half  = lane >> 4;
  const int l16   = lane & 15;
  const int hsh   = half ? 16 : 0;
  const int b     = blockIdx.y;
  const int qbase = blockIdx.x * 64;
  const size_t rowbase = (size_t)b * S_LEN;
  const float NEGINF = -__builtin_inff();

  auto stage = [&](int buf, int kb) {
    const unsigned short* kg = k + (rowbase + kb) * HDIM;
    int idx = threadIdx.x * 16;                       // 4KB / 128 thr = 32B
    async_copy16(ldsoff(&lK[buf][idx]),     kg + idx);
    async_copy16(ldsoff(&lK[buf][idx + 8]), kg + idx + 8);
    int h = threadIdx.x >> 1, c = (threadIdx.x & 1) * 16;
    const unsigned short* vg = vT + ((size_t)b * HDIM + h) * S_LEN + kb + c;
    async_copy16(ldsoff(&lV[buf][h * 32 + c]),     vg);
    async_copy16(ldsoff(&lV[buf][h * 32 + c + 8]), vg + 8);
  };

  // Q fragments (A layout, already scaled by 1/sqrt(64))
  const int mrow = qbase + wave * 16 + l16;
  const unsigned short* qp = q + (rowbase + mrow) * HDIM;
  v16bf qa[2];
  qa[0] = ld_afrag(qp + hsh,      qp + hsh + 8);
  qa[1] = ld_afrag(qp + 32 + hsh, qp + 32 + hsh + 8);

  v8f o[4];
  v8f zero = {0.f, 0.f, 0.f, 0.f, 0.f, 0.f, 0.f, 0.f};
#pragma unroll
  for (int t = 0; t < 4; ++t) o[t] = zero;
  float mrun[8], lrun[8];
#pragma unroll
  for (int r = 0; r < 8; ++r) { mrun[r] = NEGINF; lrun[r] = 0.f; }

  const int ktiles = (qbase + 64) / 32;               // causal frontier
  stage(0, 0);
  wait_async0();
  __syncthreads();

  for (int kt = 0; kt < ktiles; ++kt) {
    const int kb  = kt * 32;
    const int buf = kt & 1;
    if (kt + 1 < ktiles) stage(buf ^ 1, kb + 32);     // prefetch next tile

    // scores: two 16(q)x16(key) fragments, K-reduce over head dim
    v8f s[2];
#pragma unroll
    for (int f = 0; f < 2; ++f) {
      v8f c = zero;
#pragma unroll
      for (int hc = 0; hc < 2; ++hc) {
        const unsigned short* bp =
            lK[buf] + (f * 16 + l16) * HDIM + hc * 32 + (half ? 16 : 0);
        c = wmma_bf16(qa[hc], ld_bfrag(bp), c);
      }
      s[f] = c;
    }

    // causal mask + row-max (reduce across 16-lane groups; rows = VGPR index)
    float tmax[8];
#pragma unroll
    for (int r = 0; r < 8; ++r) {
      int row = qbase + wave * 16 + r + half * 8;
      float s0 = (kb + l16      <= row) ? s[0][r] : NEGINF;
      float s1 = (kb + 16 + l16 <= row) ? s[1][r] : NEGINF;
      s[0][r] = s0; s[1][r] = s1;
      float t = fmaxf(s0, s1);
#pragma unroll
      for (int off = 1; off < 16; off <<= 1) t = fmaxf(t, __shfl_xor(t, off, 16));
      tmax[r] = t;
    }

    // online softmax; write P (bf16) into per-wave LDS staging
    unsigned short* ps = lP[wave];
#pragma unroll
    for (int r = 0; r < 8; ++r) {
      float mnew = fmaxf(mrun[r], tmax[r]);
      float corr = __expf(mrun[r] - mnew);
      float p0 = __expf(s[0][r] - mnew);
      float p1 = __expf(s[1][r] - mnew);
      float rs = p0 + p1;
#pragma unroll
      for (int off = 1; off < 16; off <<= 1) rs += __shfl_xor(rs, off, 16);
      lrun[r] = lrun[r] * corr + rs;
      mrun[r] = mnew;
#pragma unroll
      for (int t = 0; t < 4; ++t) o[t][r] *= corr;
      int m = r + half * 8;
      ps[m * 32 + l16]      = bf16bits(p0);
      ps[m * 32 + 16 + l16] = bf16bits(p1);
    }
    asm volatile("s_wait_dscnt 0x0" ::: "memory");    // wave-local LDS RAW fence

    // O += P @ V  (P as A fragment, V column tiles from lV)
    v16bf pa = ld_afrag(ps + l16 * 32 + hsh, ps + l16 * 32 + hsh + 8);
#pragma unroll
    for (int t = 0; t < 4; ++t) {
      const unsigned short* bp = lV[buf] + (t * 16 + l16) * 32 + (half ? 16 : 0);
      o[t] = wmma_bf16(pa, ld_bfrag(bp), o[t]);
    }

    if (kt + 1 < ktiles) {                            // commit prefetched tile
      wait_async0();
      __syncthreads();
    }
  }

  // epilogue: O / l
#pragma unroll
  for (int r = 0; r < 8; ++r) {
    float inv = 1.0f / lrun[r];
    int row = qbase + wave * 16 + r + half * 8;
#pragma unroll
    for (int t = 0; t < 4; ++t)
      out[(rowbase + row) * HDIM + t * 16 + l16] = o[t][r] * inv;
  }
}

// ============================================================================
extern "C" void kernel_launch(void* const* d_in, const int* in_sizes, int n_in,
                              void* d_out, int out_size, void* d_ws, size_t ws_size,
                              hipStream_t stream) {
  const float* x  = (const float*)d_in[0];
  const float* Wq = (const float*)d_in[1];
  const float* Wk = (const float*)d_in[2];
  const float* Wv = (const float*)d_in[3];
  float* out = (float*)d_out;

  char* ws = (char*)d_ws;
  unsigned short* Wt = (unsigned short*)ws;                       // 384 KB
  unsigned short* qw = (unsigned short*)(ws + 512 * 1024);        // 2 MB each
  unsigned short* kw = qw + (size_t)ROWS * HDIM;
  unsigned short* vw = kw + (size_t)ROWS * HDIM;                  // vT[B][H][S]

  wt_kernel<<<(3 * HDIM * DMODEL + 255) / 256, 256, 0, stream>>>(Wq, Wk, Wv, Wt);
  qkv_kernel<<<ROWS / 128, 256, 0, stream>>>(x, Wt, qw, kw, vw);
  flash_kernel<<<dim3(S_LEN / 64, BATCH), 128, 0, stream>>>(qw, kw, vw, out);
}
